// pointcloud_class_44040594653470
// MI455X (gfx1250) — matrix operand order repaired and checked by reading, not compiled, not run on hardware
//
#include <hip/hip_runtime.h>
#include <hip/hip_bf16.h>
#include <math.h>

#define NSTAGES 7
#define BCNT 16
#define NPTS 2048
#define CLS 40
#define KPMAX 544
#define LN_EPS 1e-3f

typedef __attribute__((ext_vector_type(16))) __bf16 v16bf;
typedef __attribute__((ext_vector_type(8)))  float  v8f;

union V16U { v16bf v; unsigned short u[16]; uint4 q[2]; };

__device__ __forceinline__ unsigned short f2bf(float x) {
  union { float f; unsigned u; } c; c.f = x;
  unsigned r = c.u + 0x7FFFu + ((c.u >> 16) & 1u);   // round-to-nearest-even
  return (unsigned short)(r >> 16);
}

__device__ __forceinline__ v8f wmma_bf16(v16bf a, v16bf b, v8f c) {
  // D = A(16x32 bf16) * B(32x16 bf16) + C(16x16 f32)
  return __builtin_amdgcn_wmma_f32_16x16x32_bf16(false, a, false, b, (short)0, c, false, false);
}

// A fragment: 16x32 bf16 tile from row-major bf16 array (LDS or global).
// ISA layout: lane m=lane&15 is row; half-wave hi selects K groups; two contiguous
// 8-element runs -> compiler emits 2x ds_load_b128.
__device__ __forceinline__ v16bf load_a_frag(const unsigned short* A, int row0, int k0, int stride) {
  int lane = threadIdx.x & 31;
  int m = lane & 15, hi = lane >> 4;
  const unsigned short* rp = A + (long)(row0 + m) * stride;
  V16U r;
#pragma unroll
  for (int e = 0; e < 16; ++e) {
    int k = k0 + e + hi * 8 + ((e >= 8) ? 8 : 0);
    r.u[e] = rp[k];
  }
  return r.v;
}

// B fragment from pre-swizzled weights: fragment (nt,kt) stored as 32 lanes x 16 bf16
// contiguous -> two unconditional global_load_b128 per fragment.
__device__ __forceinline__ v16bf load_b_packed(const unsigned short* Wp, int kt, int nt, int KT) {
  int lane = threadIdx.x & 31;
  const uint4* p = (const uint4*)(Wp + ((((long)nt * KT + kt) * 32 + lane) << 4));
  V16U r;
  r.q[0] = p[0];
  r.q[1] = p[1];
  return r.v;
}

// ---------- one-time weight swizzle: W(Ksrc x N f32, row-major) -> bf16 fragment order ----------
// dst[((nt*KT + kt)*32 + lane)*16 + e] = bf16(W[kt*32 + e + (lane>>4)*8 + (e>=8 ? 8 : 0)][nt*16 + (lane&15)])
__global__ void k_pack_w(const float* __restrict__ W, unsigned short* __restrict__ dst,
                         int Ksrc, int N, int KT) {
  long i = (long)blockIdx.x * blockDim.x + threadIdx.x;
  long total = (long)KT * 32 * N;   // == KP * N
  if (i >= total) return;
  int e    = (int)(i & 15);
  int lane = (int)((i >> 4) & 31);
  long f   = i >> 9;
  int kt = (int)(f % KT), nt = (int)(f / KT);
  int k = kt * 32 + e + (lane >> 4) * 8 + ((e >= 8) ? 8 : 0);
  int n = nt * 16 + (lane & 15);
  float v = (k < Ksrc) ? W[(long)k * N + n] : 0.f;
  dst[i] = f2bf(v);
}

// ---------------- init MLP: 3 -> 32 (relu,LN) -> 64 (relu,LN) -----------------
__global__ void k_init_mlp(const float* __restrict__ xyz,
    const float* __restrict__ w1, const float* __restrict__ b1,
    const float* __restrict__ g0, const float* __restrict__ e0,
    const float* __restrict__ w2, const float* __restrict__ b2,
    const float* __restrict__ g1, const float* __restrict__ e1,
    float* __restrict__ f, int total) {
  int i = blockIdx.x * blockDim.x + threadIdx.x;
  if (i >= total) return;
  float x0 = xyz[i*3+0], x1 = xyz[i*3+1], x2 = xyz[i*3+2];
  float h[32]; float s = 0.f, q = 0.f;
#pragma unroll
  for (int j = 0; j < 32; ++j) {
    float a = b1[j] + x0*w1[0*32+j] + x1*w1[1*32+j] + x2*w1[2*32+j];
    a = fmaxf(a, 0.f); h[j] = a; s += a; q += a*a;
  }
  float mean = s/32.f, inv = rsqrtf(q/32.f - mean*mean + LN_EPS);
#pragma unroll
  for (int j = 0; j < 32; ++j) h[j] = (h[j]-mean)*inv*g0[j] + e0[j];
  float h2[64]; s = 0.f; q = 0.f;
  for (int j = 0; j < 64; ++j) {
    float a = b2[j];
    for (int k = 0; k < 32; ++k) a += h[k]*w2[k*64+j];
    a = fmaxf(a, 0.f); h2[j] = a; s += a; q += a*a;
  }
  mean = s/64.f; inv = rsqrtf(q/64.f - mean*mean + LN_EPS);
  for (int j = 0; j < 64; ++j) f[(long)i*64+j] = (h2[j]-mean)*inv*g1[j] + e1[j];
}

// ---------------- brute-force kNN (top-32 smallest sq-dist) -----------------
__global__ void k_knn(const float* __restrict__ xyz_q, const float* __restrict__ xyz_s,
                      int* __restrict__ idx, int Mq, int Ms, int total) {
  int i = blockIdx.x * blockDim.x + threadIdx.x;
  if (i >= total) return;
  int b = i / Mq;
  const float* qp = xyz_q + (long)i*3;
  float qx = qp[0], qy = qp[1], qz = qp[2];
  const float* sp = xyz_s + (long)b*Ms*3;
  float bd[32]; int bi[32];
#pragma unroll
  for (int k = 0; k < 32; ++k) { bd[k] = 3.4e38f; bi[k] = 0; }
  for (int j = 0; j < Ms; ++j) {
    float dx = sp[j*3+0]-qx, dy = sp[j*3+1]-qy, dz = sp[j*3+2]-qz;
    float d = dx*dx+dy*dy+dz*dz;
    if (d < bd[31]) {
      int p = 31;
      while (p > 0 && bd[p-1] > d) { bd[p] = bd[p-1]; bi[p] = bi[p-1]; --p; }
      bd[p] = d; bi[p] = j;
    }
  }
#pragma unroll
  for (int k = 0; k < 32; ++k) idx[(long)i*32+k] = bi[k];
}

// ---------------- compact xyz slice: first Mdst points per batch -----------------
__global__ void k_slice_xyz(const float* __restrict__ src, float* __restrict__ dst,
                            int Msrc, int Mdst, int total) {
  int i = blockIdx.x * blockDim.x + threadIdx.x;
  if (i >= total) return;
  int c = i % 3, t = i / 3, b = t / Mdst, m = t % Mdst;
  dst[i] = src[((long)b*Msrc + m)*3 + c];
}

// ------------- fused group + 3-head attention (WMMA QKV) per (b,m) group -------------
// wqp/wkp/wvp: pre-swizzled bf16 weights, 3 heads consecutive, per-head block = KP*dout.
__global__ __launch_bounds__(128) void k_group_attn(
    const float* __restrict__ xyz_q, const float* __restrict__ xyz_s,
    const float* __restrict__ feat, const int* __restrict__ knn,
    int Mq, int Ms, int C, int S, int local_mode,
    const unsigned short* __restrict__ wqp, const float* __restrict__ bq,
    const unsigned short* __restrict__ wkp, const float* __restrict__ bk,
    const unsigned short* __restrict__ wvp, const float* __restrict__ bv,
    const float* __restrict__ wo, const float* __restrict__ bo,
    const float* __restrict__ lng, const float* __restrict__ lnb,
    int dout,
    float* __restrict__ la, float* __restrict__ glob) {
  __shared__ unsigned short sA[32 * KPMAX];   // 34.8 KB bf16 local tile
  __shared__ float s_qm[1024];
  __shared__ float s_oacc[3072];
  __shared__ float s_o[1024];
  __shared__ float s_score[32];
  __shared__ float s_att[32];
  __shared__ int   s_idx[32];
  __shared__ float s_red[2];

  const int tid = threadIdx.x, lane = tid & 31, wid = tid >> 5;
  const int b = blockIdx.x / Mq, m = blockIdx.x % Mq;
  const int din = 3 + C;
  const int KP = (din + 31) & ~31;
  const int RT = S >> 4, NT = dout >> 4, KT = KP >> 5;
  const long hstride = (long)KP * dout;   // per-head packed weight block

  if (tid < S) s_idx[tid] = knn ? knn[((long)b*Mq + m)*S + tid] : tid;
  for (int i = tid; i < 3*dout; i += 128) s_oacc[i] = 0.f;
  __syncthreads();

  // build centered tile + glob max (one channel per thread)
  for (int c = tid; c < KP; c += 128) {
    if (c < din) {
      float vals[32]; float mx = -3.4e38f, sum = 0.f;
      for (int s = 0; s < S; ++s) {
        int j = s_idx[s];
        float v;
        if (c < 3) v = xyz_s[((long)b*Ms + j)*3 + c] - (local_mode ? xyz_q[((long)b*Mq + m)*3 + c] : 0.f);
        else       v = feat[((long)b*Ms + j)*C + (c-3)];
        vals[s] = v; sum += v; mx = fmaxf(mx, v);
      }
      float mean = local_mode ? sum / (float)S : 0.f;
      for (int s = 0; s < S; ++s) sA[s*KP + c] = f2bf(vals[s] - mean);
      for (int s = S; s < 32; ++s) sA[s*KP + c] = 0;
      if (glob) glob[((long)b*Mq + m)*din + c] = mx;
    } else {
      for (int s = 0; s < 32; ++s) sA[s*KP + c] = 0;
    }
  }
  __syncthreads();

  const float scale = rsqrtf((float)dout);
  for (int h = 0; h < 3; ++h) {
    const unsigned short* Wq = wqp + h*hstride; const float* Bq = bq + h*dout;
    const unsigned short* Wk = wkp + h*hstride; const float* Bk = bk + h*dout;
    const unsigned short* Wv = wvp + h*hstride; const float* Bv = bv + h*dout;
    (void)Bv;
    for (int i = tid; i < dout; i += 128) s_qm[i] = 0.f;
    if (tid < 32) s_score[tid] = 0.f;
    __syncthreads();

    // ---- Q pass: column sums -> qm ----
    for (int t = wid; t < RT*NT; t += 4) {
      int rt = t % RT, nt = t / RT;
      v8f acc = {0.f,0.f,0.f,0.f,0.f,0.f,0.f,0.f};
      for (int kt = 0; kt < KT; ++kt)
        acc = wmma_bf16(load_a_frag(sA, rt*16, kt*32, KP), load_b_packed(Wq, kt, nt, KT), acc);
      float cs = 0.f;
#pragma unroll
      for (int j = 0; j < 8; ++j) cs += acc[j];
      cs += __shfl_xor(cs, 16, 32);
      if ((lane >> 4) == 0) atomicAdd(&s_qm[nt*16 + (lane & 15)], cs);
    }
    __syncthreads();
    for (int i = tid; i < dout; i += 128) s_qm[i] = s_qm[i] / (float)S + Bq[i];
    __syncthreads();

    // ---- K pass: score[s] = sum_n (k_raw + bk) * qm ----
    for (int t = wid; t < RT*NT; t += 4) {
      int rt = t % RT, nt = t / RT;
      v8f acc = {0.f,0.f,0.f,0.f,0.f,0.f,0.f,0.f};
      for (int kt = 0; kt < KT; ++kt)
        acc = wmma_bf16(load_a_frag(sA, rt*16, kt*32, KP), load_b_packed(Wk, kt, nt, KT), acc);
      int n = nt*16 + (lane & 15), hi = lane >> 4;
      float qmn = s_qm[n], bkn = Bk[n];
      float part[8];
#pragma unroll
      for (int j = 0; j < 8; ++j) part[j] = (acc[j] + bkn) * qmn;
#pragma unroll
      for (int mask = 1; mask < 16; mask <<= 1)
#pragma unroll
        for (int j = 0; j < 8; ++j) part[j] += __shfl_xor(part[j], mask, 32);
      if ((lane & 15) == 0)
#pragma unroll
        for (int j = 0; j < 8; ++j) atomicAdd(&s_score[rt*16 + j + 8*hi], part[j]);
    }
    __syncthreads();

    // ---- softmax over S samples (wave 0) ----
    if (tid < 32) {
      float sc = (tid < S) ? s_score[tid] * scale : -3.4e38f;
      float mx = sc;
#pragma unroll
      for (int mask = 1; mask < 32; mask <<= 1) mx = fmaxf(mx, __shfl_xor(mx, mask, 32));
      float e = (tid < S) ? __expf(sc - mx) : 0.f;
      float se = e;
#pragma unroll
      for (int mask = 1; mask < 32; mask <<= 1) se += __shfl_xor(se, mask, 32);
      if (tid < S) s_att[tid] = e / se;
    }
    __syncthreads();

    // ---- V pass: out[n] += sum_s att[s]*v_raw[s,n] ----
    for (int t = wid; t < RT*NT; t += 4) {
      int rt = t % RT, nt = t / RT;
      v8f acc = {0.f,0.f,0.f,0.f,0.f,0.f,0.f,0.f};
      for (int kt = 0; kt < KT; ++kt)
        acc = wmma_bf16(load_a_frag(sA, rt*16, kt*32, KP), load_b_packed(Wv, kt, nt, KT), acc);
      int n = nt*16 + (lane & 15), hi = lane >> 4;
      float cs = 0.f;
#pragma unroll
      for (int j = 0; j < 8; ++j) cs += s_att[rt*16 + j + 8*hi] * acc[j];
      cs += __shfl_xor(cs, 16, 32);
      if (hi == 0) atomicAdd(&s_oacc[h*dout + n], cs);
    }
    __syncthreads();
  }

  // add v-bias (softmax weights sum to 1)
  for (int i = tid; i < 3*dout; i += 128) s_oacc[i] += bv[(i/dout)*dout + (i%dout)];
  __syncthreads();

  // output projection (3*dout -> dout) + relu
  int K2 = 3*dout;
  for (int d = tid; d < dout; d += 128) {
    float a = bo[d];
    for (int k = 0; k < K2; ++k) a += s_oacc[k] * wo[(long)k*dout + d];
    s_o[d] = fmaxf(a, 0.f);
  }
  if (tid == 0) { s_red[0] = 0.f; s_red[1] = 0.f; }
  __syncthreads();
  float ls = 0.f, lq = 0.f;
  for (int d = tid; d < dout; d += 128) { float v = s_o[d]; ls += v; lq += v*v; }
  atomicAdd(&s_red[0], ls); atomicAdd(&s_red[1], lq);
  __syncthreads();
  float mean = s_red[0] / dout, var = s_red[1] / dout - mean*mean, inv = rsqrtf(var + LN_EPS);
  for (int d = tid; d < dout; d += 128)
    la[((long)b*Mq + m)*dout + d] = (s_o[d]-mean)*inv*lng[d] + lnb[d];
}

// ---------------- pack concat([A(R x Ca), Bm(R x Cb)]) -> zero-padded bf16 (RP x KP) -------------
__global__ void k_pack2(const float* __restrict__ Aa, int Ca, const float* __restrict__ Ab, int Cb,
                        unsigned short* __restrict__ out, int R, int RP, int KP) {
  long i = (long)blockIdx.x * blockDim.x + threadIdx.x;
  long total = (long)RP * KP;
  if (i >= total) return;
  int r = (int)(i / KP), c = (int)(i % KP);
  float v = 0.f;
  if (r < R) {
    if (c < Ca) v = Aa[(long)r*Ca + c];
    else if (c < Ca + Cb) v = Ab[(long)r*Cb + (c - Ca)];
  }
  out[i] = f2bf(v);
}

// ---------------- WMMA GEMM: out(RxN) = relu(A_bf16(RPxKP) @ Wp + bias) -------------
__global__ __launch_bounds__(128) void k_gemm_bf16(
    const unsigned short* __restrict__ A, const unsigned short* __restrict__ Wp,
    const float* __restrict__ bias, float* __restrict__ out,
    int R, int KP, int N, int relu) {
  int wid = threadIdx.x >> 5, lane = threadIdx.x & 31;
  int tiles_r = (R + 15) >> 4, tiles_n = N >> 4;
  int t = blockIdx.x * 4 + wid;
  if (t >= tiles_r * tiles_n) return;      // uniform per wave -> EXEC all-1 for wmma
  int rt = t % tiles_r, nt = t / tiles_r;
  v8f acc = {0.f,0.f,0.f,0.f,0.f,0.f,0.f,0.f};
  int KT = KP >> 5;
  for (int kt = 0; kt < KT; ++kt) {
    if (kt + 1 < KT) __builtin_prefetch(Wp + (((long)nt*KT + kt + 1) << 9), 0, 1);  // global_prefetch_b8
    acc = wmma_bf16(load_a_frag(A, rt*16, kt*32, KP), load_b_packed(Wp, kt, nt, KT), acc);
  }
  int n = nt*16 + (lane & 15), hi = lane >> 4;
  float bvv = bias ? bias[n] : 0.f;
#pragma unroll
  for (int j = 0; j < 8; ++j) {
    int r = rt*16 + j + 8*hi;
    if (r < R) {
      float v = acc[j] + bvv;
      if (relu) v = fmaxf(v, 0.f);
      out[(long)r*N + n] = v;
    }
  }
}

// ---------------- row-wise LayerNorm in place -----------------
__global__ __launch_bounds__(128) void k_ln_rows(float* __restrict__ x,
    const float* __restrict__ g, const float* __restrict__ bta, int N) {
  __shared__ float red[2];
  long r = blockIdx.x;
  if (threadIdx.x == 0) { red[0] = 0.f; red[1] = 0.f; }
  __syncthreads();
  float s = 0.f, q = 0.f;
  for (int c = threadIdx.x; c < N; c += blockDim.x) { float v = x[r*N + c]; s += v; q += v*v; }
  atomicAdd(&red[0], s); atomicAdd(&red[1], q);
  __syncthreads();
  float mean = red[0]/N, var = red[1]/N - mean*mean, inv = rsqrtf(var + LN_EPS);
  for (int c = threadIdx.x; c < N; c += blockDim.x)
    x[r*N + c] = (x[r*N + c] - mean)*inv*g[c] + bta[c];
}

// ---------------- classification head: 1024 -> 256 (LN) -> 256 (LN) -> 40 softmax ----------
__global__ __launch_bounds__(256) void k_head(
    const float* __restrict__ o,
    const float* w3, const float* b3, const float* g2, const float* e2,
    const float* w4, const float* b4, const float* g3, const float* e3,
    const float* w5, const float* b5, float* __restrict__ out) {
  __shared__ float s_in[1024], s_a[256], s_b[256], red[2], s_logit[CLS];
  int b = blockIdx.x, tid = threadIdx.x;
  for (int i = tid; i < 1024; i += 256) s_in[i] = o[(long)b*1024 + i];
  __syncthreads();
  { float a = b3[tid]; for (int k = 0; k < 1024; ++k) a += s_in[k]*w3[k*256+tid]; s_a[tid] = fmaxf(a,0.f); }
  if (tid == 0) { red[0]=0.f; red[1]=0.f; }
  __syncthreads();
  atomicAdd(&red[0], s_a[tid]); atomicAdd(&red[1], s_a[tid]*s_a[tid]);
  __syncthreads();
  { float mean = red[0]/256.f, var = red[1]/256.f - mean*mean, inv = rsqrtf(var + LN_EPS);
    s_a[tid] = (s_a[tid]-mean)*inv*g2[tid] + e2[tid]; }
  __syncthreads();
  { float a = b4[tid]; for (int k = 0; k < 256; ++k) a += s_a[k]*w4[k*256+tid]; s_b[tid] = fmaxf(a,0.f); }
  __syncthreads();
  if (tid == 0) { red[0]=0.f; red[1]=0.f; }
  __syncthreads();
  atomicAdd(&red[0], s_b[tid]); atomicAdd(&red[1], s_b[tid]*s_b[tid]);
  __syncthreads();
  { float mean = red[0]/256.f, var = red[1]/256.f - mean*mean, inv = rsqrtf(var + LN_EPS);
    s_b[tid] = (s_b[tid]-mean)*inv*g3[tid] + e3[tid]; }
  __syncthreads();
  if (tid < CLS) { float a = b5[tid]; for (int k = 0; k < 256; ++k) a += s_b[k]*w5[k*CLS+tid]; s_logit[tid] = a; }
  __syncthreads();
  if (tid == 0) {
    float mx = -3.4e38f;
    for (int c = 0; c < CLS; ++c) mx = fmaxf(mx, s_logit[c]);
    float se = 0.f;
    for (int c = 0; c < CLS; ++c) se += __expf(s_logit[c] - mx);
    for (int c = 0; c < CLS; ++c) out[(long)b*CLS + c] = __expf(s_logit[c] - mx) / se;
  }
}

extern "C" void kernel_launch(void* const* d_in, const int* in_sizes, int n_in,
                              void* d_out, int out_size, void* d_ws, size_t ws_size,
                              hipStream_t stream) {
  (void)in_sizes; (void)n_in; (void)out_size; (void)ws_size;
  static const int LAT[8] = {64,64,128,128,256,256,512,1024};
  static const int CIN[7] = {64,64,64,128,128,256,256};

  const float* xyz = (const float*)d_in[0];
  int ix = 1;
  auto nx = [&]() { return (const float*)d_in[ix++]; };
  const float *d1w=nx(),*d1b=nx(),*ln0g=nx(),*ln0b=nx();
  const float *d2w=nx(),*d2b=nx(),*ln1g=nx(),*ln1b=nx();
  struct AttnP { const float *wq,*bq,*wk,*bk,*wv,*bv,*wo,*bo,*lg,*lb; };
  struct DnsP  { const float *w,*b; };
  struct LnP   { const float *g,*b; };
  AttnP attn2[NSTAGES], attn1[NSTAGES], attnF;
  DnsP dn2[NSTAGES], dn1[NSTAGES];
  LnP  n1[NSTAGES], n2[NSTAGES];
  auto rdA = [&](AttnP& a){ a.wq=nx();a.bq=nx();a.wk=nx();a.bk=nx();a.wv=nx();a.bv=nx();a.wo=nx();a.bo=nx();a.lg=nx();a.lb=nx(); };
  for (int i=0;i<NSTAGES;++i) rdA(attn2[i]);
  for (int i=0;i<NSTAGES;++i){ dn2[i].w=nx(); dn2[i].b=nx(); }
  for (int i=0;i<NSTAGES;++i){ n1[i].g=nx(); n1[i].b=nx(); }
  for (int i=0;i<NSTAGES;++i) rdA(attn1[i]);
  for (int i=0;i<NSTAGES;++i){ dn1[i].w=nx(); dn1[i].b=nx(); }
  for (int i=0;i<NSTAGES;++i){ n2[i].g=nx(); n2[i].b=nx(); }
  rdA(attnF);
  const float *d3w=nx(),*d3b=nx(),*ln2g=nx(),*ln2b=nx();
  const float *d4w=nx(),*d4b=nx(),*ln3g=nx(),*ln3b=nx();
  const float *d5w=nx(),*d5b=nx();

  // workspace carve
  char* base = (char*)d_ws; size_t off = 0;
  auto carve = [&](size_t bytes) { off = (off + 255) & ~(size_t)255; void* p = base + off; off += bytes; return p; };
  float* fA  = (float*)carve((size_t)BCNT*NPTS*64*sizeof(float));   // stage features
  float* fB  = (float*)carve((size_t)BCNT*NPTS*64*sizeof(float));   // intermediate features
  int*   idb = (int*)  carve((size_t)BCNT*NPTS*32*sizeof(int));
  float* lab = (float*)carve((size_t)BCNT*NPTS*64*sizeof(float));   // attention outputs
  float* glb = (float*)carve((size_t)BCNT*NPTS*80*sizeof(float));   // group max features
  unsigned short* pkb = (unsigned short*)carve((size_t)8*1024*1024*sizeof(unsigned short));
  float* xyzA = (float*)carve((size_t)BCNT*NPTS*3*sizeof(float));
  float* xyzB = (float*)carve((size_t)BCNT*NPTS*3*sizeof(float));
  float* laF  = (float*)carve((size_t)BCNT*1024*sizeof(float));
  unsigned short* wb = (unsigned short*)carve((size_t)20*1024*1024*sizeof(unsigned short));

  // ---- one-time weight swizzle to WMMA-B fragment order (bf16, zero-padded) ----
  size_t woff = 0;
  auto packW = [&](const float* W, int Ksrc, int N) {
    int KP = (Ksrc + 31) & ~31;
    unsigned short* dst = wb + woff;
    long total = (long)KP * N;
    k_pack_w<<<(int)((total + 255) / 256), 256, 0, stream>>>(W, dst, Ksrc, N, KP >> 5);
    woff += (size_t)total;
    return (const unsigned short*)dst;
  };
  // attention QKV: 3 heads packed consecutively (per-head block = KP*dout)
  struct AttnW { const unsigned short *wq, *wk, *wv; };
  auto packAttn = [&](const AttnP& a, int din, int dout) {
    AttnW w;
    w.wq = packW(a.wq + 0, din, dout); packW(a.wq + (long)din*dout, din, dout); packW(a.wq + 2L*din*dout, din, dout);
    w.wk = packW(a.wk + 0, din, dout); packW(a.wk + (long)din*dout, din, dout); packW(a.wk + 2L*din*dout, din, dout);
    w.wv = packW(a.wv + 0, din, dout); packW(a.wv + (long)din*dout, din, dout); packW(a.wv + 2L*din*dout, din, dout);
    return w;
  };
  AttnW attn2w[NSTAGES], attn1w[NSTAGES], attnFw;
  const unsigned short *dn2w[NSTAGES], *dn1w[NSTAGES];
  for (int i = 0; i < NSTAGES; ++i) {
    attn2w[i] = packAttn(attn2[i], 3 + CIN[i], LAT[i]);
    attn1w[i] = packAttn(attn1[i], 3 + LAT[i], LAT[i]);
    dn2w[i] = packW(dn2[i].w, LAT[i] + 3 + CIN[i], LAT[i]);
    dn1w[i] = packW(dn1[i].w, 2*LAT[i] + 3, LAT[i]);
  }
  attnFw = packAttn(attnF, 3 + LAT[NSTAGES-1], LAT[NSTAGES]);

  // ---- pipeline ----
  int totalPts = BCNT*NPTS;
  k_init_mlp<<<(totalPts+127)/128, 128, 0, stream>>>(xyz, d1w,d1b,ln0g,ln0b,d2w,d2b,ln1g,ln1b, fA, totalPts);

  const float* oxyz = xyz;
  int M = NPTS;
  for (int i = 0; i < NSTAGES; ++i) {
    int C = CIN[i], D = LAT[i];
    int din1 = 3 + C;
    int nq = BCNT * M;
    // ---- first half: self grouping on old_xyz ----
    k_knn<<<(nq+127)/128,128,0,stream>>>(oxyz, oxyz, idb, M, M, nq);
    k_group_attn<<<nq,128,0,stream>>>(oxyz, oxyz, fA, idb, M, M, C, 32, 1,
        attn2w[i].wq, attn2[i].bq, attn2w[i].wk, attn2[i].bk, attn2w[i].wv, attn2[i].bv,
        attn2[i].wo, attn2[i].bo, attn2[i].lg, attn2[i].lb, D, lab, glb);
    int K1 = D + din1, KP1 = (K1 + 31) & ~31;
    int R1 = nq, RP1 = (R1 + 15) & ~15;
    long tot1 = (long)RP1 * KP1;
    k_pack2<<<(int)((tot1+255)/256),256,0,stream>>>(lab, D, glb, din1, pkb, R1, RP1, KP1);
    int tiles1 = (RP1/16) * (D/16);
    k_gemm_bf16<<<(tiles1+3)/4,128,0,stream>>>(pkb, dn2w[i], dn2[i].b, fB, R1, KP1, D, 1);
    k_ln_rows<<<R1,128,0,stream>>>(fB, n1[i].g, n1[i].b, D);
    // ---- second half: decimated queries against old points ----
    int Mh = M/2, nq2 = BCNT*Mh;
    float* nxyz = (i & 1) ? xyzB : xyzA;
    int tsl = BCNT*Mh*3;
    k_slice_xyz<<<(tsl+127)/128,128,0,stream>>>(oxyz, nxyz, M, Mh, tsl);
    k_knn<<<(nq2+127)/128,128,0,stream>>>(nxyz, oxyz, idb, Mh, M, nq2);
    int din2 = 3 + D;
    k_group_attn<<<nq2,128,0,stream>>>(nxyz, oxyz, fB, idb, Mh, M, D, 32, 1,
        attn1w[i].wq, attn1[i].bq, attn1w[i].wk, attn1[i].bk, attn1w[i].wv, attn1[i].bv,
        attn1[i].wo, attn1[i].bo, attn1[i].lg, attn1[i].lb, D, lab, glb);
    int K2 = D + din2, KP2 = (K2 + 31) & ~31;
    int R2 = nq2, RP2 = (R2 + 15) & ~15;
    long tot2 = (long)RP2 * KP2;
    k_pack2<<<(int)((tot2+255)/256),256,0,stream>>>(lab, D, glb, din2, pkb, R2, RP2, KP2);
    int tiles2 = (RP2/16) * (D/16);
    k_gemm_bf16<<<(tiles2+3)/4,128,0,stream>>>(pkb, dn1w[i], dn1[i].b, fA, R2, KP2, D, 1);
    k_ln_rows<<<R2,128,0,stream>>>(fA, n2[i].g, n2[i].b, D);
    oxyz = nxyz; M = Mh;
  }

  // final attention over all 16 remaining points per batch (raw concat, no centering)
  k_group_attn<<<BCNT,128,0,stream>>>(oxyz, oxyz, fA, (const int*)nullptr,
      1, M, LAT[NSTAGES-1], 16, 0,
      attnFw.wq, attnF.bq, attnFw.wk, attnF.bk, attnFw.wv, attnF.bv,
      attnF.wo, attnF.bo, attnF.lg, attnF.lb, LAT[NSTAGES], laF, (float*)nullptr);

  k_head<<<BCNT,256,0,stream>>>(laF, d3w,d3b,ln2g,ln2b, d4w,d4b,ln3g,ln3b, d5w,d5b, (float*)d_out);
}
